// CapsuleLayer_14551349199211
// MI455X (gfx1250) — matrix operand order repaired and checked by reading, not compiled, not run on hardware
//
#include <hip/hip_runtime.h>

// Problem constants (match reference)
#define BB   32
#define RR   2048
#define CIN  64
#define COUT 64
#define KK   16

typedef __attribute__((ext_vector_type(16))) __bf16 v16bf;
typedef __attribute__((ext_vector_type(8)))  __bf16 v8bf;
typedef __attribute__((ext_vector_type(4)))  __bf16 v4bf;
typedef __attribute__((ext_vector_type(2)))  __bf16 v2bf;
typedef __attribute__((ext_vector_type(8)))  float  v8f;

__device__ __forceinline__ __bf16 f2bf(float f) { return (__bf16)f; }  // v_cvt_(pk_)bf16_f32
__device__ __forceinline__ float bf16_to_f32(unsigned short h) {
  return __uint_as_float(((unsigned)h) << 16);
}

// wave32 shuffle reductions (no LDS, no barriers)
__device__ __forceinline__ float wave_reduce_add(float v) {
#pragma unroll
  for (int m = 16; m > 0; m >>= 1) v += __shfl_xor(v, m, 32);
  return v;
}
__device__ __forceinline__ float wave_reduce_max(float v) {
#pragma unroll
  for (int m = 16; m > 0; m >>= 1) v = fmaxf(v, __shfl_xor(v, m, 32));
  return v;
}

// ---------------------------------------------------------------------------
// Kernel 1: priors[k,b,r,o] = sum_c x[b,r,c] * W[k,r,c,o], stored as bf16.
// One block (256 thr = 8 waves, wave32) per r; loop over k so W is read once
// (512 MB -> single HBM pass). Each wave owns one 16x16 (b x o) output tile
// and chains two v_wmma_f32_16x16x32_bf16 over c=64. W staging is
// double-buffered: k+1 global loads issue before the k WMMAs, cvt+LDS stores
// after, one barrier per k.
// ---------------------------------------------------------------------------
__global__ void __launch_bounds__(256)
capsule_priors_gemm(const float* __restrict__ x,
                    const float* __restrict__ w,
                    __bf16* __restrict__ priors) {
  const int r    = blockIdx.x;
  const int t    = threadIdx.x;
  const int lane = t & 31;          // wave32
  const int wv   = t >> 5;          // 8 waves
  const int mt   = wv & 1;          // M tile (b rows 0..15 / 16..31)
  const int nt   = wv >> 1;         // N tile (o cols, 4 tiles of 16)

  // stride 72 halves = 144 B rows: 16B-aligned rows and bank-spread gathers
  __shared__ __bf16 sA[32 * 72];       // x tile   [b][c]
  __shared__ __bf16 sB[2][64 * 72];    // W tile^T [o][c], double-buffered

  // ---- stage x[0:32, r, 0:64] -> sA: 2 packed cvts + ds_store_b64 each ----
#pragma unroll
  for (int i = 0; i < 2; ++i) {
    int f4 = t + i * 256;                    // 512 float4 total
    int b  = f4 >> 4;
    int c4 = (f4 & 15) << 2;
    const float4 v = *(const float4*)(x + (size_t)b * (RR * CIN) +
                                      (size_t)r * CIN + c4);
    v4bf p = { f2bf(v.x), f2bf(v.y), f2bf(v.z), f2bf(v.w) };
    *(v4bf*)&sA[b * 72 + c4] = p;
  }

  // staging geometry for W: task f covers c = 2*(f>>4), c+1 and o = (f&15)*4..+3
  const int cp0 = t >> 4,         o40 = (t & 15) << 2;          // task 0
  const int cp1 = (t + 256) >> 4, o41 = ((t + 256) & 15) << 2;  // task 1

  // ---- prologue: stage W[k=0] into sB[0] ----
  {
    const float* W0 = w + ((size_t)r << 12);
    const float* b0 = W0 + (size_t)(cp0 * 2) * 64 + o40;
    const float* b1 = W0 + (size_t)(cp1 * 2) * 64 + o41;
    float4 a0 = *(const float4*)(b0);
    float4 a1 = *(const float4*)(b0 + 64);
    float4 a2 = *(const float4*)(b1);
    float4 a3 = *(const float4*)(b1 + 64);
    const float *pa0 = (const float*)&a0, *pa1 = (const float*)&a1;
    const float *pa2 = (const float*)&a2, *pa3 = (const float*)&a3;
#pragma unroll
    for (int j = 0; j < 4; ++j) {
      v2bf p0 = { f2bf(pa0[j]), f2bf(pa1[j]) };   // (o40+j, c), (o40+j, c+1)
      *(v2bf*)&sB[0][(o40 + j) * 72 + cp0 * 2] = p0;
      v2bf p1 = { f2bf(pa2[j]), f2bf(pa3[j]) };
      *(v2bf*)&sB[0][(o41 + j) * 72 + cp1 * 2] = p1;
    }
  }
  __syncthreads();

  // ---- build A fragments once (ISA 16-bit A 16x32 layout, wave32) ----
  // lane<16 : halves 0..7 = K 0..7,  halves 8..15 = K 16..23 (M = lane)
  // lane>=16: halves 0..7 = K 8..15, halves 8..15 = K 24..31 (M = lane-16)
  const int mrow = mt * 16 + (lane & 15);
  const int hiA  = (lane & 16) ? 8 : 0;
  v16bf afrag[2];
#pragma unroll
  for (int kb = 0; kb < 2; ++kb) {
    const int ka0 = kb * 32 + hiA;
    const v8bf* p0 = (const v8bf*)&sA[mrow * 72 + ka0];
    const v8bf* p1 = (const v8bf*)&sA[mrow * 72 + ka0 + 16];
#pragma unroll
    for (int i = 0; i < 8; ++i) { afrag[kb][i] = (*p0)[i]; afrag[kb][i + 8] = (*p1)[i]; }
  }

  // B layout: lane n -> col o = n%16; halves e -> K as A (e<8: K=e, else K=e+8; +8 for lanes 16..31)
  const int ocol = nt * 16 + (lane & 15);
  const int hiB  = (lane & 16) ? 16 : 0;

  for (int k = 0; k < KK; ++k) {
    const int cur = k & 1;
    const bool have_next = (k + 1 < KK);

    // issue next W tile's global loads early (hide under WMMA + stores)
    float4 n0 = {}, n1 = {}, n2 = {}, n3 = {};
    if (have_next) {
      const float* Wn = w + (((size_t)(k + 1) * RR + r) << 12);
      const float* b0 = Wn + (size_t)(cp0 * 2) * 64 + o40;
      const float* b1 = Wn + (size_t)(cp1 * 2) * 64 + o41;
      n0 = *(const float4*)(b0);
      n1 = *(const float4*)(b0 + 64);
      n2 = *(const float4*)(b1);
      n3 = *(const float4*)(b1 + 64);
    }
    if (k + 2 < KK && t < 128) {      // global_prefetch_b8 for k+2 (16 KB tile)
      const float* W2 = w + (((size_t)(k + 2) * RR + r) << 12);
      __builtin_prefetch(W2 + t * 32, 0, 0);
    }

    // ---- compute k from sB[cur] ----
    v8f acc = {};
#pragma unroll
    for (int kb = 0; kb < 2; ++kb) {
      v16bf bfrag;
      const v8bf* q0 = (const v8bf*)&sB[cur][ocol * 72 + kb * 32 + hiA];
      const v8bf* q1 = (const v8bf*)&sB[cur][ocol * 72 + kb * 32 + hiA + 16];
#pragma unroll
      for (int i = 0; i < 8; ++i) { bfrag[i] = (*q0)[i]; bfrag[i + 8] = (*q1)[i]; }
      acc = __builtin_amdgcn_wmma_f32_16x16x32_bf16(false, afrag[kb], false, bfrag,
                                                    (short)0, acc, false, false);
    }

    // C/D layout: VGPR v -> M = v (+8 for lanes 16..31); N = lane%16
#pragma unroll
    for (int v = 0; v < 8; ++v) {
      int bb = mt * 16 + v + ((lane & 16) ? 8 : 0);
      size_t idx = ((((size_t)k * BB + bb) * RR) + r) * COUT + ocol;
      priors[idx] = f2bf(acc[v]);
    }

    // ---- stage next tile into sB[cur^1]: packed cvt + ds_store_b32 ----
    if (have_next) {
      const int nb = cur ^ 1;
      const float *pa0 = (const float*)&n0, *pa1 = (const float*)&n1;
      const float *pa2 = (const float*)&n2, *pa3 = (const float*)&n3;
#pragma unroll
      for (int j = 0; j < 4; ++j) {
        v2bf p0 = { f2bf(pa0[j]), f2bf(pa1[j]) };
        *(v2bf*)&sB[nb][(o40 + j) * 72 + cp0 * 2] = p0;
        v2bf p1 = { f2bf(pa2[j]), f2bf(pa3[j]) };
        *(v2bf*)&sB[nb][(o41 + j) * 72 + cp1 * 2] = p1;
      }
    }
    __syncthreads();   // staging of k+1 visible; sB[cur] reads complete
  }
}

// ---------------------------------------------------------------------------
// Kernel 2: full dynamic routing for one (k,b) per block. Logits / exp
// weights live in LDS; the block's 256 KB priors slice is re-read from L2
// (whole 128 MB priors array ~fits the 192 MB L2 across resident blocks).
// Reductions use wave32 shuffles + one 8-entry cross-wave combine.
// ---------------------------------------------------------------------------
__global__ void __launch_bounds__(256)
capsule_routing(const unsigned short* __restrict__ priors,
                float* __restrict__ out) {
  const int kb   = blockIdx.x;               // k*32 + b
  const int t    = threadIdx.x;
  const int lane = t & 31;
  const int wv   = t >> 5;
  const unsigned short* P = priors + (size_t)kb * (RR * COUT);

  __shared__ float sl[RR];                   // logits
  __shared__ float se[RR];                   // exp(logits - max)
  __shared__ float spart[16][COUT];          // per-group partial sums
  __shared__ float sred[8];                  // per-wave scalars
  __shared__ float vout[COUT];               // current output vector
  __shared__ float sfac[2];                  // [0]=squash factor, [1]=1/Z

  const int og = (t & 15) << 2;              // 4 consecutive o per thread
  const int rg = t >> 4;                     // 16 r groups

  auto weighted_sum_squash = [&](bool useW, float invZ) {
    float s0 = 0.f, s1 = 0.f, s2 = 0.f, s3 = 0.f;
    for (int r = rg; r < RR; r += 16) {
      float wt = useW ? se[r] : 1.0f;
      const uint2 u = *(const uint2*)(P + (size_t)r * COUT + og);
      s0 += wt * bf16_to_f32((unsigned short)(u.x & 0xFFFFu));
      s1 += wt * bf16_to_f32((unsigned short)(u.x >> 16));
      s2 += wt * bf16_to_f32((unsigned short)(u.y & 0xFFFFu));
      s3 += wt * bf16_to_f32((unsigned short)(u.y >> 16));
    }
    spart[rg][og + 0] = s0;
    spart[rg][og + 1] = s1;
    spart[rg][og + 2] = s2;
    spart[rg][og + 3] = s3;
    __syncthreads();
    if (t < COUT) {
      float s = 0.f;
#pragma unroll
      for (int g = 0; g < 16; ++g) s += spart[g][t];
      vout[t] = s * invZ;
    }
    __syncthreads();
    float sq = (t < COUT) ? vout[t] * vout[t] : 0.f;
    sq = wave_reduce_add(sq);
    if (lane == 0) sred[wv] = sq;
    __syncthreads();
    if (t == 0) {
      float s = sred[0] + sred[1] + sred[2] + sred[3] +
                sred[4] + sred[5] + sred[6] + sred[7];
      sfac[0] = (s / (1.0f + s)) * rsqrtf(s);
    }
    __syncthreads();
    if (t < COUT) vout[t] *= sfac[0];
    __syncthreads();
  };

  auto update_logits_softmax = [&](bool accumulate) {
    for (int r = t; r < RR; r += 256) {
      const uint4* row = (const uint4*)(P + (size_t)r * COUT);
      float acc = 0.f;
#pragma unroll
      for (int j = 0; j < 8; ++j) {
        uint4 u = row[j];
        const float* vo = &vout[j * 8];
        acc += bf16_to_f32((unsigned short)(u.x & 0xFFFFu)) * vo[0]
             + bf16_to_f32((unsigned short)(u.x >> 16))     * vo[1]
             + bf16_to_f32((unsigned short)(u.y & 0xFFFFu)) * vo[2]
             + bf16_to_f32((unsigned short)(u.y >> 16))     * vo[3]
             + bf16_to_f32((unsigned short)(u.z & 0xFFFFu)) * vo[4]
             + bf16_to_f32((unsigned short)(u.z >> 16))     * vo[5]
             + bf16_to_f32((unsigned short)(u.w & 0xFFFFu)) * vo[6]
             + bf16_to_f32((unsigned short)(u.w >> 16))     * vo[7];
      }
      sl[r] = accumulate ? (sl[r] + acc) : acc;
    }
    __syncthreads();
    // block max via wave shuffles
    float m = -3.0e38f;
    for (int r = t; r < RR; r += 256) m = fmaxf(m, sl[r]);
    m = wave_reduce_max(m);
    if (lane == 0) sred[wv] = m;
    __syncthreads();
    m = sred[0];
#pragma unroll
    for (int g = 1; g < 8; ++g) m = fmaxf(m, sred[g]);
    __syncthreads();                         // sred reused below
    // exp + block sum
    float z = 0.f;
    for (int r = t; r < RR; r += 256) {
      float ex = __expf(sl[r] - m);
      se[r] = ex;
      z += ex;
    }
    z = wave_reduce_add(z);
    if (lane == 0) sred[wv] = z;
    __syncthreads();
    if (t == 0) {
      float Z = sred[0] + sred[1] + sred[2] + sred[3] +
                sred[4] + sred[5] + sred[6] + sred[7];
      sfac[1] = 1.0f / Z;
    }
    __syncthreads();
  };

  // iter 0: uniform probs = 1/R
  weighted_sum_squash(false, 1.0f / (float)RR);
  // iter 1
  update_logits_softmax(false);
  weighted_sum_squash(true, sfac[1]);
  // iter 2
  update_logits_softmax(true);
  weighted_sum_squash(true, sfac[1]);

  if (t < COUT) out[(size_t)kb * COUT + t] = vout[t];
}

// ---------------------------------------------------------------------------
extern "C" void kernel_launch(void* const* d_in, const int* in_sizes, int n_in,
                              void* d_out, int out_size, void* d_ws, size_t ws_size,
                              hipStream_t stream) {
  (void)in_sizes; (void)n_in; (void)out_size; (void)ws_size;
  const float* x = (const float*)d_in[0];                 // [32,2048,64]
  const float* w = (const float*)d_in[1];                 // [16,2048,64,64]
  float* out = (float*)d_out;                             // [16,32,64]
  // priors bf16 [16,32,2048,64] = 128 MB scratch in d_ws
  __bf16* priors = (__bf16*)d_ws;

  capsule_priors_gemm<<<dim3(RR), dim3(256), 0, stream>>>(x, w, priors);
  capsule_routing<<<dim3(KK * BB), dim3(256), 0, stream>>>(
      (const unsigned short*)d_ws, out);
}
